// RedundantHRRSelfAttention_58634893525658
// MI455X (gfx1250) — compile-verified, matrix operand order checked by reading, not run on hardware
//
#include <hip/hip_runtime.h>
#include <hip/hip_bf16.h>

// ---------------------------------------------------------------------------
// RedundantHRRSelfAttention for MI455X (gfx1250, wave32, WMMA).
//   B=2, T=2048, D=1024, C=10, F=513.
// Strategy: every dense op (projections, rfft, irfft, output proj) is a
// bf16 WMMA GEMM (v_wmma_f32_16x16x32_bf16) with a double-buffered
// GLOBAL_LOAD_ASYNC_TO_LDS_B128 pipeline (ASYNCcnt); the causal HRR scan is
// a chunked two-phase prefix scan in f32 VALU over L2-resident spectra.
// Workspace requirement: ~122 MB.
// ---------------------------------------------------------------------------

typedef __bf16 bf16_t;
typedef __attribute__((ext_vector_type(16))) __bf16 v16bf;
typedef __attribute__((ext_vector_type(8)))  __bf16 bf16x8;
typedef __attribute__((ext_vector_type(8)))  float  v8f;
typedef int i32x4_vs __attribute__((vector_size(16)));   // matches builtin V4i

#define Dd     1024
#define Mrows  4096          // B*T
#define Cc     10
#define Fb     513           // rfft bins
#define Fp     576           // padded bins per (re|im) half -> 1152 cols total
#define LDF    (2*Fp)        // 1152 (re block | im block)
#define NCH    8             // scan chunks over T
#define CHLEN  256           // 2048 / NCH
#define TWO_PI_OVER_N 6.1359231515e-03f   // 2*pi/1024

// CDNA5 async global->LDS path (ASYNCcnt). Falls back to the synchronous
// load+ds_store pipeline if this toolchain lacks the builtins.
#if defined(__AMDGCN__) && \
    __has_builtin(__builtin_amdgcn_global_load_async_to_lds_b128) && \
    __has_builtin(__builtin_amdgcn_s_wait_asynccnt)
#define USE_ASYNC_LDS 1
#else
#define USE_ASYNC_LDS 0
#endif

#if USE_ASYNC_LDS
__device__ __forceinline__ void async_cp_b128(const bf16_t* g, bf16_t* l)
{
    __builtin_amdgcn_global_load_async_to_lds_b128(
        (__attribute__((address_space(1))) i32x4_vs*)(void*)g,
        (__attribute__((address_space(3))) i32x4_vs*)(void*)l, 0, 0);
}
#endif

// Fragment loads per ISA 16-bit A/B layout + 8 WMMA issues for one K=32 step.
//   lanes 0-15 : K 0-7 (v0-3), K 16-23 (v4-7)
//   lanes 16-31: K 8-15,        K 24-31
__device__ __forceinline__ void frag_mma(const bf16_t* __restrict__ pA,
                                         const bf16_t* __restrict__ pB,
                                         int wm, int wn, int rlo, int kh,
                                         v8f acc[4][2])
{
    v16bf fa[4], fb[2];
#pragma unroll
    for (int tm = 0; tm < 4; ++tm) {
        const int r = wm * 64 + tm * 16 + rlo;
        bf16x8 lo = *(const bf16x8*)(&pA[r * 40 + kh]);
        bf16x8 hi = *(const bf16x8*)(&pA[r * 40 + kh + 16]);
#pragma unroll
        for (int e = 0; e < 8; ++e) { fa[tm][e] = lo[e]; fa[tm][8 + e] = hi[e]; }
    }
#pragma unroll
    for (int tn = 0; tn < 2; ++tn) {
        const int r = wn * 32 + tn * 16 + rlo;
        bf16x8 lo = *(const bf16x8*)(&pB[r * 40 + kh]);
        bf16x8 hi = *(const bf16x8*)(&pB[r * 40 + kh + 16]);
#pragma unroll
        for (int e = 0; e < 8; ++e) { fb[tn][e] = lo[e]; fb[tn][8 + e] = hi[e]; }
    }
#pragma unroll
    for (int tm = 0; tm < 4; ++tm)
#pragma unroll
        for (int tn = 0; tn < 2; ++tn)
            acc[tm][tn] = __builtin_amdgcn_wmma_f32_16x16x32_bf16(
                false, fa[tm], false, fb[tn], (short)0, acc[tm][tn],
                false, false);
}

// ---------------------------------------------------------------------------
// Generic NT GEMM:  C[M x N] = A[M x K] * B[N x K]^T   (both K-contiguous)
// Block: 256 thr (8 waves as 2Mx4N), tile 128x128, wave tile 64x32, K-step 32.
// Double-buffered async global->LDS staging. Output f32 (Cf) or bf16 (Cb).
// ---------------------------------------------------------------------------
__global__ __launch_bounds__(256)
void wmma_gemm_nt(const bf16_t* __restrict__ A, const bf16_t* __restrict__ B,
                  float* __restrict__ Cf, bf16_t* __restrict__ Cb,
                  int K, int lda, int ldb, int ldc,
                  long aBatch, long cBatch)
{
    // 40-half row pitch (80 B): 16 fragment lanes land in 16 distinct banks.
    __shared__ bf16_t sA[2][128 * 40];
    __shared__ bf16_t sB[2][128 * 40];

    const int  tid  = threadIdx.x;
    const int  lane = tid & 31;
    const int  wave = tid >> 5;
    const int  wm   = wave >> 2;        // 0..1  (M)
    const int  wn   = wave & 3;         // 0..3  (N)
    const long gM0  = (long)blockIdx.y * 128;
    const long gN0  = (long)blockIdx.x * 128;

    A += (long)blockIdx.z * aBatch;

    v8f acc[4][2] = {};

    const int rlo = lane & 15;          // fragment row/col within 16
    const int kh  = (lane >> 4) * 8;    // K-half select per ISA 16-bit layout

    // Per-thread staging coordinates: 2 x b128 per matrix per K-step.
    const int r0 = tid >> 2;            // rows r0 and r0+64
    const int s0 = (tid & 3) * 8;       // 8-half segment within the 32-wide slab

    const int nsteps = K >> 5;

#if USE_ASYNC_LDS
    // ---- stage 0 ----
#pragma unroll
    for (int it = 0; it < 2; ++it) {
        const int row = r0 + it * 64;
        async_cp_b128(A + (gM0 + row) * lda + s0, &sA[0][row * 40 + s0]);
        async_cp_b128(B + (gN0 + row) * ldb + s0, &sB[0][row * 40 + s0]);
    }
    for (int s = 0; s < nsteps; ++s) {
        const int buf = s & 1;
        __syncthreads();                 // all reads of buf^1 (stage s-1) done
        if (s + 1 < nsteps) {
            const int k0n = (s + 1) << 5;
#pragma unroll
            for (int it = 0; it < 2; ++it) {
                const int row = r0 + it * 64;
                async_cp_b128(A + (gM0 + row) * lda + k0n + s0,
                              &sA[buf ^ 1][row * 40 + s0]);
                async_cp_b128(B + (gN0 + row) * ldb + k0n + s0,
                              &sB[buf ^ 1][row * 40 + s0]);
            }
            if (s + 2 < nsteps) {        // pull stage s+2 toward the WGP
                __builtin_prefetch(A + (gM0 + r0) * lda + k0n + 32 + s0, 0, 3);
                __builtin_prefetch(B + (gN0 + r0) * ldb + k0n + 32 + s0, 0, 3);
            }
            __builtin_amdgcn_s_wait_asynccnt(4);   // stage s landed; s+1 in flight
        } else {
            __builtin_amdgcn_s_wait_asynccnt(0);
        }
        __syncthreads();                 // stage s visible to all waves
        frag_mma(sA[buf], sB[buf], wm, wn, rlo, kh, acc);
    }
#else
    // ---- synchronous fallback: single-buffer global_load -> ds_store ----
    for (int s = 0; s < nsteps; ++s) {
        const int k0 = s << 5;
        __syncthreads();
#pragma unroll
        for (int it = 0; it < 2; ++it) {
            const int row = r0 + it * 64;
            *(bf16x8*)(&sA[0][row * 40 + s0]) =
                *(const bf16x8*)(A + (gM0 + row) * lda + k0 + s0);
            *(bf16x8*)(&sB[0][row * 40 + s0]) =
                *(const bf16x8*)(B + (gN0 + row) * ldb + k0 + s0);
        }
        if (s + 1 < nsteps) {
            __builtin_prefetch(A + (gM0 + r0) * lda + k0 + 32 + s0, 0, 3);
            __builtin_prefetch(B + (gN0 + r0) * ldb + k0 + 32 + s0, 0, 3);
        }
        __syncthreads();
        frag_mma(sA[0], sB[0], wm, wn, rlo, kh, acc);
    }
#endif

    // Epilogue: C layout = lanes 0-15 -> N=lane, M=r; lanes 16-31 -> M=8+r.
    const int mo = (lane >> 4) * 8;
    if (Cf) {
        Cf += (long)blockIdx.z * cBatch;
#pragma unroll
        for (int tm = 0; tm < 4; ++tm)
#pragma unroll
            for (int tn = 0; tn < 2; ++tn) {
                const long gm = gM0 + wm * 64 + tm * 16 + mo;
                const long gn = gN0 + wn * 32 + tn * 16 + rlo;
#pragma unroll
                for (int r = 0; r < 8; ++r)
                    Cf[(gm + r) * ldc + gn] = acc[tm][tn][r];
            }
    } else {
        Cb += (long)blockIdx.z * cBatch;
#pragma unroll
        for (int tm = 0; tm < 4; ++tm)
#pragma unroll
            for (int tn = 0; tn < 2; ++tn) {
                const long gm = gM0 + wm * 64 + tm * 16 + mo;
                const long gn = gN0 + wn * 32 + tn * 16 + rlo;
#pragma unroll
                for (int r = 0; r < 8; ++r)
                    Cb[(gm + r) * ldc + gn] = (bf16_t)acc[tm][tn][r];
            }
    }
}

// ---------------------------------------------------------------------------
// f32 -> bf16 conversion of x, fused [Wq;Wk;Wv], Wo; zero padded r buffer.
// ---------------------------------------------------------------------------
__global__ void convert_inputs(const float* __restrict__ x,
                               const float* __restrict__ Wq,
                               const float* __restrict__ Wk,
                               const float* __restrict__ Wv,
                               const float* __restrict__ Wo,
                               bf16_t* __restrict__ xb, bf16_t* __restrict__ wqkv,
                               bf16_t* __restrict__ wob, bf16_t* __restrict__ rb)
{
    const long i = (long)blockIdx.x * blockDim.x + threadIdx.x;
    if (i < (long)Mrows * Dd) xb[i] = (bf16_t)x[i];
    if (i < (long)Dd * Dd)    wob[i] = (bf16_t)Wo[i];
    if (i < 3L * Dd * Dd) {
        const long w = i / ((long)Dd * Dd);
        const long j = i - w * (long)Dd * Dd;
        const float* src = (w == 0) ? Wq : (w == 1) ? Wk : Wv;
        wqkv[i] = (bf16_t)src[j];
    }
    if (i < (long)Mrows * LDF) rb[i] = (bf16_t)0.0f;
}

// ---------------------------------------------------------------------------
// rfft as matrix: dftB is (LDF x D) row-major; row n<Fp: cos(2pi k f/N),
// row n>=Fp: -sin(...). Rows for f>=513 are zero (padding).
// ---------------------------------------------------------------------------
__global__ void gen_dft(bf16_t* __restrict__ dftB)
{
    const long i = (long)blockIdx.x * blockDim.x + threadIdx.x;
    if (i >= (long)LDF * Dd) return;
    const int n = (int)(i / Dd);
    const int k = (int)(i % Dd);
    const int f = (n < Fp) ? n : n - Fp;
    float v = 0.0f;
    if (f < Fb) {
        const int   m   = (k * f) & (Dd - 1);          // exact phase mod N
        const float ang = (float)m * TWO_PI_OVER_N;
        v = (n < Fp) ? __cosf(ang) : -__sinf(ang);
    }
    dftB[i] = (bf16_t)v;
}

// irfft as matrix: idftB is (D x LDF); col j<Fp: w_f cos /N, col j>=Fp: -w_f sin /N
// with w_0 = w_{N/2} = 1, else 2 (hermitian fold); padded cols zero.
__global__ void gen_idft(bf16_t* __restrict__ idftB)
{
    const long i = (long)blockIdx.x * blockDim.x + threadIdx.x;
    if (i >= (long)Dd * LDF) return;
    const int n = (int)(i / LDF);                      // output sample k
    const int j = (int)(i % LDF);
    const int f = (j < Fp) ? j : j - Fp;
    float v = 0.0f;
    if (f < Fb) {
        const int   m   = (n * f) & (Dd - 1);
        const float ang = (float)m * TWO_PI_OVER_N;
        const float w   = (f == 0 || f == Dd / 2) ? 1.0f : 2.0f;
        v = ((j < Fp) ? __cosf(ang) : -__sinf(ang)) * (w * (1.0f / (float)Dd));
    }
    idftB[i] = (bf16_t)v;
}

// ---------------------------------------------------------------------------
// Causal HRR scan, phase 1: per (b,f,chunk) compute chunk-total of
// a_t[c] = fk[b,t,perm[c,f]] * fv[b,t,f]  (complex, C=10 accumulators).
// ---------------------------------------------------------------------------
__global__ void scan_phase1(const float* __restrict__ fftk,
                            const float* __restrict__ fftv,
                            const int*   __restrict__ perms,
                            float*       __restrict__ csums)
{
    const int tid   = blockIdx.x * blockDim.x + threadIdx.x;
    const int total = 2 * Fb * NCH;
    if (tid >= total) return;
    const int chunk = tid % NCH;
    const int p     = tid / NCH;
    const int f     = p % Fb;
    const int b     = p / Fb;

    int pidx[Cc];
#pragma unroll
    for (int c = 0; c < Cc; ++c) pidx[c] = perms[c * Fb + f];

    float aR[Cc] = {}, aI[Cc] = {};
    const int t0 = chunk * CHLEN;
    for (int t = t0; t < t0 + CHLEN; ++t) {
        const long m  = (long)(b * 2048 + t) * LDF;
        const float vr = fftv[m + f];
        const float vi = fftv[m + Fp + f];
#pragma unroll
        for (int c = 0; c < Cc; ++c) {
            const float kr = fftk[m + pidx[c]];
            const float ki = fftk[m + Fp + pidx[c]];
            aR[c] += kr * vr - ki * vi;
            aI[c] += kr * vi + ki * vr;
        }
    }
    float* out = csums + (long)tid * (2 * Cc);
#pragma unroll
    for (int c = 0; c < Cc; ++c) { out[2 * c] = aR[c]; out[2 * c + 1] = aI[c]; }
}

// Phase 2: seed with exclusive prefix of chunk totals, replay local cumsum,
// unbind with conj(fq), mean over copies, write bf16 [re(576)|im(576)] rows.
__global__ void scan_phase2(const float* __restrict__ fftq,
                            const float* __restrict__ fftk,
                            const float* __restrict__ fftv,
                            const int*   __restrict__ perms,
                            const float* __restrict__ csums,
                            bf16_t*      __restrict__ rb)
{
    const int tid   = blockIdx.x * blockDim.x + threadIdx.x;
    const int total = 2 * Fb * NCH;
    if (tid >= total) return;
    const int chunk = tid % NCH;
    const int p     = tid / NCH;
    const int f     = p % Fb;
    const int b     = p / Fb;

    int pidx[Cc];
#pragma unroll
    for (int c = 0; c < Cc; ++c) pidx[c] = perms[c * Fb + f];

    float aR[Cc] = {}, aI[Cc] = {};
    const float* cs = csums + (long)(p * NCH) * (2 * Cc);
    for (int ch = 0; ch < chunk; ++ch)
#pragma unroll
        for (int c = 0; c < Cc; ++c) {
            aR[c] += cs[ch * 2 * Cc + 2 * c];
            aI[c] += cs[ch * 2 * Cc + 2 * c + 1];
        }

    const int t0 = chunk * CHLEN;
    for (int t = t0; t < t0 + CHLEN; ++t) {
        const long m  = (long)(b * 2048 + t) * LDF;
        const float vr = fftv[m + f];
        const float vi = fftv[m + Fp + f];
        float rR = 0.0f, rI = 0.0f;
#pragma unroll
        for (int c = 0; c < Cc; ++c) {
            const float kr = fftk[m + pidx[c]];
            const float ki = fftk[m + Fp + pidx[c]];
            aR[c] += kr * vr - ki * vi;
            aI[c] += kr * vi + ki * vr;
            const float qr = fftq[m + pidx[c]];
            const float qi = fftq[m + Fp + pidx[c]];
            rR += aR[c] * qr + aI[c] * qi;   // kv * conj(q)
            rI += aI[c] * qr - aR[c] * qi;
        }
        rb[m + f]      = (bf16_t)(rR * 0.1f);
        rb[m + Fp + f] = (bf16_t)(rI * 0.1f);
    }
}

// ---------------------------------------------------------------------------
// Launch pipeline.
// ---------------------------------------------------------------------------
extern "C" void kernel_launch(void* const* d_in, const int* in_sizes, int n_in,
                              void* d_out, int out_size, void* d_ws, size_t ws_size,
                              hipStream_t stream)
{
    (void)in_sizes; (void)n_in; (void)out_size; (void)ws_size;
    const float* x     = (const float*)d_in[0];
    const float* Wq    = (const float*)d_in[1];
    const float* Wk    = (const float*)d_in[2];
    const float* Wv    = (const float*)d_in[3];
    const float* Wo    = (const float*)d_in[4];
    const int*   perms = (const int*)d_in[5];
    float*       out   = (float*)d_out;

    char* ws = (char*)d_ws;
    bf16_t* xb    = (bf16_t*)(ws + 0);           //  8.0 MB  x bf16
    bf16_t* wqkv  = (bf16_t*)(ws + 8388608);     //  6.0 MB  [Wq;Wk;Wv] bf16
    bf16_t* wob   = (bf16_t*)(ws + 14680064);    //  2.0 MB  Wo bf16
    bf16_t* qkv   = (bf16_t*)(ws + 16777216);    // 24.0 MB  (4096 x 3072) bf16
    float*  fft   = (float*) (ws + 41943040);    // 54.0 MB  (3 x 4096 x 1152) f32
    bf16_t* dftB  = (bf16_t*)(ws + 98566144);    //  2.25 MB (1152 x 1024) bf16
    bf16_t* idftB = (bf16_t*)(ws + 100925440);   //  2.25 MB (1024 x 1152) bf16
    bf16_t* rb    = (bf16_t*)(ws + 103284736);   //  9.0 MB  (4096 x 1152) bf16
    bf16_t* vhat  = (bf16_t*)(ws + 112721920);   //  8.0 MB  (4096 x 1024) bf16
    float*  csums = (float*) (ws + 121110528);   //  0.63 MB scan chunk totals

    // 1) dtype conversion + r-buffer zero (covers all ranges; max idx 4096*1152)
    convert_inputs<<<18432, 256, 0, stream>>>(x, Wq, Wk, Wv, Wo,
                                              xb, wqkv, wob, rb);
    // 2) DFT / iDFT coefficient matrices (on device, deterministic)
    gen_dft <<<4608, 256, 0, stream>>>(dftB);
    gen_idft<<<4608, 256, 0, stream>>>(idftB);

    // 3) fused QKV projection: (4096x1024) x (3072x1024)^T -> bf16 (4096x3072)
    wmma_gemm_nt<<<dim3(24, 32, 1), 256, 0, stream>>>(
        xb, wqkv, nullptr, qkv, 1024, 1024, 1024, 3072, 0, 0);

    // 4) rfft via GEMM, batched over q,k,v (grid.z): A offset 0/1024/2048 into
    //    qkv rows, out (4096 x 1152) f32 per signal.
    wmma_gemm_nt<<<dim3(9, 32, 3), 256, 0, stream>>>(
        qkv, dftB, fft, nullptr, 1024, 3072, 1024, LDF,
        1024L, (long)Mrows * LDF);

    // 5) causal HRR scan (two-phase chunked prefix, 8208 independent lanes)
    const float* fftq = fft;
    const float* fftk = fft + 1L * Mrows * LDF;
    const float* fftv = fft + 2L * Mrows * LDF;
    scan_phase1<<<33, 256, 0, stream>>>(fftk, fftv, perms, csums);
    scan_phase2<<<33, 256, 0, stream>>>(fftq, fftk, fftv, perms, csums, rb);

    // 6) irfft via GEMM: (4096x1152) x (1024x1152)^T -> bf16 (4096x1024)
    wmma_gemm_nt<<<dim3(8, 32, 1), 256, 0, stream>>>(
        rb, idftB, nullptr, vhat, 1152, LDF, LDF, 1024, 0, 0);

    // 7) output projection: (4096x1024) x (1024x1024)^T -> f32 d_out
    wmma_gemm_nt<<<dim3(8, 32, 1), 256, 0, stream>>>(
        vhat, wob, out, nullptr, 1024, 1024, 1024, 1024, 0, 0);
}
